// MPNN_8864812499459
// MI455X (gfx1250) — matrix operand order repaired and checked by reading, not compiled, not run on hardware
//
#include <hip/hip_runtime.h>
#include <hip/hip_bf16.h>

typedef __attribute__((ext_vector_type(16))) _Float16 v16h;
typedef __attribute__((ext_vector_type(8)))  _Float16 h8;
typedef __attribute__((ext_vector_type(4)))  _Float16 h4;
typedef __attribute__((ext_vector_type(8)))  float    v8f;

#define NN 1024
#define DD 64
#define NFEAT 4
#define ROWLEN (NFEAT + 2*NN)   // 2052

// ---------------------------------------------------------------------------
// Kernel 1: per-row prep.  One wave per row i of batch b.
//   t[b,i]     = sum_j adj*e            (wave reduction)
//   adjh       = (f16) adj              (row major, WMMA A operand)
//   base[b,i,d]= nf@W1 + t * (relu(W4)@W3)[d]
//   emb1       = relu(base)  -> embT0 (f16, transposed [d][n])
// ---------------------------------------------------------------------------
__global__ __launch_bounds__(256) void mpnn_prep(
    const float* __restrict__ state,
    const float* __restrict__ W1,   // (4,64)
    const float* __restrict__ W3,   // (64,64)
    const float* __restrict__ W4,   // (1,64)
    float* __restrict__ base,       // [B][N][D]
    _Float16* __restrict__ adjh,    // [B][N][N]
    _Float16* __restrict__ embT0)   // [B][D][N]
{
    const int b    = blockIdx.x >> 7;          // 128 blocks per batch
    const int i    = ((blockIdx.x & 127) << 3) + (threadIdx.x >> 5);
    const int lane = threadIdx.x & 31;

    const float* row  = state + ((size_t)b * NN + i) * ROWLEN;
    const float* adjp = row + NFEAT;
    const float* ep   = row + NFEAT + NN;
    _Float16*    ah   = adjh + ((size_t)b * NN + i) * NN;

    float tsum = 0.f;
    for (int c = 0; c < NN; c += 128) {
        const int j = c + lane * 4;
        const float4 a4 = *(const float4*)(adjp + j);
        const float4 e4 = *(const float4*)(ep + j);
        tsum += a4.x * e4.x + a4.y * e4.y + a4.z * e4.z + a4.w * e4.w;
        h4 hv = { (_Float16)a4.x, (_Float16)a4.y, (_Float16)a4.z, (_Float16)a4.w };
        *(h4*)(ah + j) = hv;
    }
    for (int off = 16; off; off >>= 1) tsum += __shfl_down(tsum, off, 32);
    tsum = __shfl(tsum, 0, 32);

    // each lane handles dims d = lane and lane+32
    const int d = lane;
    float x1a = 0.f, x1b = 0.f;
    #pragma unroll
    for (int f = 0; f < NFEAT; ++f) {
        const float nf = row[f];
        x1a += nf * W1[f * DD + d];
        x1b += nf * W1[f * DD + d + 32];
    }
    float va = 0.f, vb = 0.f;
    #pragma unroll 8
    for (int c = 0; c < DD; ++c) {
        const float w4 = fmaxf(W4[c], 0.f);       // relu(W4) folded in (e >= 0)
        va += w4 * W3[c * DD + d];
        vb += w4 * W3[c * DD + d + 32];
    }
    const float ba = x1a + tsum * va;
    const float bb = x1b + tsum * vb;

    float* bp = base + ((size_t)b * NN + i) * DD;
    bp[d]      = ba;
    bp[d + 32] = bb;

    _Float16* et = embT0 + (size_t)b * DD * NN;
    et[(size_t)d        * NN + i] = (_Float16)fmaxf(ba, 0.f);
    et[(size_t)(d + 32) * NN + i] = (_Float16)fmaxf(bb, 0.f);
}

// ---------------------------------------------------------------------------
// Kernel 2: one MPNN iteration  emb_out = relu(base + (adj @ emb_in) @ W2)
// Block = 256 threads = 8 waves, covers 32 rows x 64 cols.
// Each wave: 16x16 tile of M = adj@emb via v_wmma_f32_16x16x32_f16, K = 1024.
// ---------------------------------------------------------------------------
__global__ __launch_bounds__(256) void mpnn_iter(
    const _Float16* __restrict__ adjh,    // [B][N][N]
    const _Float16* __restrict__ embT_in, // [B][D][N]
    const float* __restrict__ base,       // [B][N][D]
    const float* __restrict__ W2,         // (64,64)
    float* __restrict__ emb_out,          // [B][N][D]
    _Float16* __restrict__ embT_out)      // [B][D][N]
{
    __shared__ float Mlds[32][68];        // padded to dodge bank conflicts
    __shared__ float W2s[DD * DD];

    const int b  = blockIdx.x >> 5;       // 32 row-groups per batch
    const int rg = blockIdx.x & 31;
    const int t  = threadIdx.x;

    for (int idx = t; idx < DD * DD; idx += 256) W2s[idx] = W2[idx];

    const int wave = t >> 5;
    const int lane = t & 31;
    const int rt = wave >> 2;             // row tile 0..1
    const int ct = wave & 3;              // col tile 0..3
    const int i0 = rg * 32 + rt * 16;
    const int c0 = ct * 16;

    const _Float16* Abase = adjh   + ((size_t)b * NN + i0) * NN;
    const _Float16* Bbase = embT_in + (size_t)b * DD * NN;

    const int m  = lane & 15;             // A row / B column within tile
    const int kh = lane >> 4;             // lane half

    v8f acc = {};
    for (int k = 0; k < NN; k += 32) {
        // A fragment (16x32 f16): two contiguous 8-half chunks per lane
        const _Float16* ap = Abase + (size_t)m * NN + k + kh * 16;
        h8 alo = *(const h8*)ap;
        h8 ahi = *(const h8*)(ap + 8);
        v16h A = __builtin_shufflevector(alo, ahi,
                    0,1,2,3,4,5,6,7,8,9,10,11,12,13,14,15);
        // B fragment (32x16 f16): 16 contiguous halves along K (transposed emb)
        const _Float16* bp = Bbase + (size_t)(c0 + m) * NN + k + kh * 16;
        v16h Bt = *(const v16h*)bp;
        if (k + 32 < NN) __builtin_prefetch(ap + 32, 0, 0);
        acc = __builtin_amdgcn_wmma_f32_16x16x32_f16(
                  false, A, false, Bt, (short)0, acc, false, false);
    }

    // scatter C/D fragment to LDS: lane n = lane&15 ; rows mb..mb+7
    const int nn2 = lane & 15;
    const int mb  = (lane >> 4) * 8;
    #pragma unroll
    for (int r = 0; r < 8; ++r)
        Mlds[rt * 16 + mb + r][c0 + nn2] = acc[r];
    __syncthreads();

    // phase 2: H = M @ W2 (f32 VALU), out = relu(base + H)
    const int il = t >> 3;                // local row 0..31
    const int d0 = (t & 7) * 8;           // 8 consecutive dims
    float av[8] = {0,0,0,0,0,0,0,0};
    for (int k = 0; k < DD; ++k) {
        const float mval = Mlds[il][k];
        const float* w = &W2s[k * DD + d0];
        #pragma unroll
        for (int r = 0; r < 8; ++r) av[r] += mval * w[r];
    }
    const int gi = rg * 32 + il;
    const float* bp2 = base    + ((size_t)b * NN + gi) * DD + d0;
    float*       eo  = emb_out + ((size_t)b * NN + gi) * DD + d0;
    _Float16*    et  = embT_out + (size_t)b * DD * NN;
    #pragma unroll
    for (int r = 0; r < 8; ++r) {
        const float v = fmaxf(bp2[r] + av[r], 0.f);
        eo[r] = v;
        et[(size_t)(d0 + r) * NN + gi] = (_Float16)v;
    }
}

// ---------------------------------------------------------------------------
// Kernel 3a: per-batch  qb[b] = sum_d relu((emb.sum(0) @ W6)[d]) * W5[d]
// ---------------------------------------------------------------------------
__global__ __launch_bounds__(256) void mpnn_readout_a(
    const float* __restrict__ emb,   // [B][N][D]
    const float* __restrict__ W6,    // (64,64)
    const float* __restrict__ W5,    // (128,1)
    float* __restrict__ qb)          // [B]
{
    __shared__ float sred[256];
    __shared__ float se[DD];
    __shared__ float qred[DD];
    const int b = blockIdx.x;
    const int t = threadIdx.x;
    const int d = t & 63;
    const int chunk = t >> 6;        // 4 chunks of 256 rows

    float s = 0.f;
    const float* ep = emb + (size_t)b * NN * DD + d;
    for (int i = chunk * 256; i < chunk * 256 + 256; ++i)
        s += ep[(size_t)i * DD];
    sred[t] = s;
    __syncthreads();
    if (t < DD) se[t] = sred[t] + sred[t + 64] + sred[t + 128] + sred[t + 192];
    __syncthreads();
    if (t < DD) {
        float x6 = 0.f;
        for (int c = 0; c < DD; ++c) x6 += se[c] * W6[c * DD + t];
        qred[t] = fmaxf(x6, 0.f) * W5[t];
    }
    __syncthreads();
    if (t == 0) {
        float q = 0.f;
        for (int k = 0; k < DD; ++k) q += qred[k];
        qb[b] = q;
    }
}

// ---------------------------------------------------------------------------
// Kernel 3b: per-row  q[b,i] = qb[b] + sum_d relu((emb[b,i]@W7)[d])*W5[64+d]
// One wave per row; lane handles dims (lane, lane+32).
// ---------------------------------------------------------------------------
__global__ __launch_bounds__(256) void mpnn_readout_b(
    const float* __restrict__ emb,   // [B][N][D]
    const float* __restrict__ W7,    // (64,64)
    const float* __restrict__ W5,    // (128,1)
    const float* __restrict__ qb,    // [B]
    float* __restrict__ out)         // [B][N]
{
    const int b    = blockIdx.x >> 7;
    const int i    = ((blockIdx.x & 127) << 3) + (threadIdx.x >> 5);
    const int lane = threadIdx.x & 31;
    const float* er = emb + ((size_t)b * NN + i) * DD;

    float a1 = 0.f, a2 = 0.f;
    #pragma unroll 8
    for (int c = 0; c < DD; ++c) {
        const float e = er[c];
        a1 += e * W7[c * DD + lane];
        a2 += e * W7[c * DD + lane + 32];
    }
    float s = fmaxf(a1, 0.f) * W5[DD + lane] + fmaxf(a2, 0.f) * W5[DD + lane + 32];
    for (int off = 16; off; off >>= 1) s += __shfl_down(s, off, 32);
    if (lane == 0) out[(size_t)b * NN + i] = qb[b] + s;
}

// ---------------------------------------------------------------------------
extern "C" void kernel_launch(void* const* d_in, const int* in_sizes, int n_in,
                              void* d_out, int out_size, void* d_ws, size_t ws_size,
                              hipStream_t stream) {
    (void)in_sizes; (void)n_in; (void)out_size; (void)ws_size;
    const float* state = (const float*)d_in[0];
    const float* W1    = (const float*)d_in[1];
    const float* W2    = (const float*)d_in[2];
    const float* W3    = (const float*)d_in[3];
    const float* W4    = (const float*)d_in[4];
    const float* W5    = (const float*)d_in[5];
    const float* W6    = (const float*)d_in[6];
    const float* W7    = (const float*)d_in[7];
    float* out = (float*)d_out;

    char* ws = (char*)d_ws;
    float*    base  = (float*)(ws);                                   // 1 MB
    float*    emb   = (float*)(ws + (1u << 20));                      // 1 MB
    _Float16* embA  = (_Float16*)(ws + (2u << 20));                   // 512 KB
    _Float16* embB  = (_Float16*)(ws + (2u << 20) + (512u << 10));    // 512 KB
    _Float16* adjh  = (_Float16*)(ws + (3u << 20));                   // 8 MB
    float*    qb    = (float*)(ws + (11u << 20));                     // 16 B

    // prep: base, adj->f16, emb1 = relu(base) (first of T=4 iterations)
    mpnn_prep<<<512, 256, 0, stream>>>(state, W1, W3, W4, base, adjh, embA);
    // remaining 3 iterations (ping-pong transposed f16 emb)
    mpnn_iter<<<128, 256, 0, stream>>>(adjh, embA, base, W2, emb, embB);
    mpnn_iter<<<128, 256, 0, stream>>>(adjh, embB, base, W2, emb, embA);
    mpnn_iter<<<128, 256, 0, stream>>>(adjh, embA, base, W2, emb, embB);
    // readout
    mpnn_readout_a<<<4, 256, 0, stream>>>(emb, W6, W5, qb);
    mpnn_readout_b<<<512, 256, 0, stream>>>(emb, W7, W5, qb, out);
}
